// RNN_27358941675539
// MI455X (gfx1250) — matrix operand order repaired and checked by reading, not compiled, not run on hardware
//
#include <hip/hip_runtime.h>

// gfx1250 is wave32; WMMA bf16: D(16x16 f32) = A(16x32 bf16) x B(32x16 bf16) + C
typedef __attribute__((ext_vector_type(16))) __bf16 v16bf;
typedef __attribute__((ext_vector_type(8)))  float  v8f;

union Frag {
    v16bf v;
    uint4 q[2];
};

__device__ __forceinline__ float sigmoidf_(float x) {
    return 1.0f / (1.0f + __expf(-x));
}

// ---------------------------------------------------------------------------
// Prep: convert W [512,1024] f32 (row k, col n) -> Wt [1024,512] bf16 (row n),
// zero the h double buffers and the grid-barrier counters. Runs every call so
// graph replays are deterministic.
// ---------------------------------------------------------------------------
__global__ __launch_bounds__(256)
void lstm_prep_kernel(const float* __restrict__ W0, const float* __restrict__ W1,
                      __bf16* __restrict__ Wt0, __bf16* __restrict__ Wt1,
                      __bf16* __restrict__ hz, unsigned* __restrict__ cnts)
{
    const int i = blockIdx.x * blockDim.x + threadIdx.x;   // 0 .. 524287
    const int k = i >> 10;        // 0..511  (K row of original W)
    const int n = i & 1023;       // 0..1023 (gate column)
    Wt0[(size_t)n * 512 + k] = (__bf16)W0[i];
    Wt1[(size_t)n * 512 + k] = (__bf16)W1[i];
    if (i < 4 * 32 * 256) hz[i] = (__bf16)0.0f;            // 4 h buffers
    if (i < 2) cnts[i] = 0u;                               // 2 barrier counters
}

// ---------------------------------------------------------------------------
// Persistent weight-stationary LSTM layer: 8 workgroups x 256 threads
// (8 waves). WG hb owns hidden units [hb*32, hb*32+32) -> gate columns
// {g*256 + hb*32 + u}. Each wave computes one 16-wide gate-column tile for
// both 16-row batch tiles, and holds its entire B slice (16 cols x K=512
// bf16 = 128 VGPRs) resident in registers for all T steps: weights are read
// from L2 exactly once per layer. Grid barrier serializes the recurrence.
// ---------------------------------------------------------------------------
__global__ __launch_bounds__(256, 1)
void lstm_layer_kernel(const void* __restrict__ in_seq, int in_is_f32,
                       long in_bs, long in_ts,
                       const __bf16* __restrict__ Wt,     // [1024][512] bf16
                       const float* __restrict__ bias,    // [1024] f32
                       void* __restrict__ out_seq, int out_is_f32,
                       long out_bs, long out_ts,
                       __bf16* __restrict__ hbuf,         // [2][32*256] bf16
                       unsigned* __restrict__ cnt, int T)
{
    __shared__ __bf16 sA[32 * 512];    // [batch][K] activations, 32 KB
    __shared__ float  sG[32 * 128];    // [batch][local gate col] f32, 16 KB
    __shared__ float  sBias[128];

    const int tid  = threadIdx.x;
    const int hb   = blockIdx.x;       // hidden slice 0..7
    const int wv   = tid >> 5;         // wave 0..7
    const int lane = tid & 31;
    const int l15  = lane & 15;
    const int hi   = lane >> 4;        // K-half select (contiguous k order used here)

    // wave's 16 contiguous gate columns (local cols jloc..jloc+15)
    const int jloc    = wv * 16;
    const int gate    = jloc >> 5;                 // 0..3 (i,j,f,o)
    const int colbase = gate * 256 + hb * 32 + (jloc & 31);

    if (tid < 128) {
        const int g = tid >> 5, u = tid & 31;
        sBias[tid] = bias[g * 256 + hb * 32 + u];
    }

    // ---- preload the wave's entire weight slice into registers ----
    // B 32x16 bf16 per K-chunk: lanes 0-15 -> K 0..15, lanes 16-31 -> K 16..31
    Frag breg[16];
    {
        const __bf16* bbase = Wt + (size_t)(colbase + l15) * 512 + hi * 16;
        #pragma unroll
        for (int kc = 0; kc < 16; ++kc) {
            breg[kc].q[0] = *(const uint4*)(bbase + kc * 32);
            breg[kc].q[1] = *(const uint4*)(bbase + kc * 32 + 8);
        }
    }

    // c-state in registers: thread owns the same 4 (b,u) cells every step
    float cstate[4] = {0.0f, 0.0f, 0.0f, 0.0f};
    __syncthreads();

    for (int t = 0; t < T; ++t) {
        const __bf16* hcur = hbuf + ((t     & 1) * (32 * 256));
        __bf16*       hnxt = hbuf + (((t+1) & 1) * (32 * 256));

        // ---- assemble A = [x_t | h_prev], bf16, row stride 512 ----
        for (int e = tid; e < 32 * 256; e += 256) {
            const int b = e >> 8, d = e & 255;
            float xv;
            if (in_is_f32)
                xv = ((const float*)in_seq)[(long)b * in_bs + (long)t * in_ts + d];
            else
                xv = (float)((const __bf16*)in_seq)[(long)b * in_bs + (long)t * in_ts + d];
            sA[b * 512 + d]       = (__bf16)xv;
            sA[b * 512 + 256 + d] = hcur[b * 256 + d];
        }
        __syncthreads();

        // ---- GEMM: acc[2][16x16] over K=512; B already in registers ----
        // A 16x32 bf16: lanes 0-15 -> K 0..15, lanes 16-31 -> K 16..31
        v8f acc0 = {};  // batch rows 0..15
        v8f acc1 = {};  // batch rows 16..31
        const __bf16* arow0 = &sA[l15 * 512 + hi * 16];
        const __bf16* arow1 = &sA[(16 + l15) * 512 + hi * 16];
        #pragma unroll
        for (int kc = 0; kc < 16; ++kc) {
            Frag a0, a1;
            a0.q[0] = *(const uint4*)(arow0 + kc * 32);
            a0.q[1] = *(const uint4*)(arow0 + kc * 32 + 8);
            a1.q[0] = *(const uint4*)(arow1 + kc * 32);
            a1.q[1] = *(const uint4*)(arow1 + kc * 32 + 8);
            acc0 = __builtin_amdgcn_wmma_f32_16x16x32_bf16(false, a0.v, false, breg[kc].v,
                                                           (short)0, acc0, false, false);
            acc1 = __builtin_amdgcn_wmma_f32_16x16x32_bf16(false, a1.v, false, breg[kc].v,
                                                           (short)0, acc1, false, false);
        }

        // ---- spill accumulators to LDS gates buffer (C layout: vgpr r ->
        //      M=r for lanes 0-15, M=8+r for lanes 16-31; N = lane&15) ----
        #pragma unroll
        for (int r = 0; r < 8; ++r) {
            sG[(r + hi * 8) * 128 + jloc + l15]      = acc0[r];
            sG[(16 + r + hi * 8) * 128 + jloc + l15] = acc1[r];
        }
        __syncthreads();

        // ---- elementwise LSTM update (TF BasicLSTMCell, forget_bias=1) ----
        #pragma unroll
        for (int e = 0; e < 4; ++e) {
            const int id = tid + 256 * e;          // 0..1023 -> (b,u)
            const int b = id >> 5, u = id & 31;
            const float gi = sG[b * 128 + u]        + sBias[u];
            const float gj = sG[b * 128 + 32 + u]   + sBias[32 + u];
            const float gf = sG[b * 128 + 64 + u]   + sBias[64 + u];
            const float go = sG[b * 128 + 96 + u]   + sBias[96 + u];
            const float cn = cstate[e] * sigmoidf_(gf + 1.0f)
                           + sigmoidf_(gi) * tanhf(gj);
            const float hn = tanhf(cn) * sigmoidf_(go);
            cstate[e] = cn;
            const int n = hb * 32 + u;
            hnxt[b * 256 + n] = (__bf16)hn;
            if (out_is_f32)
                ((float*)out_seq)[(long)b * out_bs + (long)t * out_ts + n] = hn;
            else
                ((__bf16*)out_seq)[(long)b * out_bs + (long)t * out_ts + n] = (__bf16)hn;
        }

        // ---- device-wide barrier: publish h_next, wait for all 8 WGs ----
        __threadfence();
        __syncthreads();
        if (tid == 0) {
            __hip_atomic_fetch_add(cnt, 1u, __ATOMIC_ACQ_REL, __HIP_MEMORY_SCOPE_AGENT);
            const unsigned target = 8u * (unsigned)(t + 1);
            while (__hip_atomic_load(cnt, __ATOMIC_ACQUIRE, __HIP_MEMORY_SCOPE_AGENT) < target) {
                __builtin_amdgcn_s_sleep(1);
            }
        }
        __syncthreads();
    }
}

// ---------------------------------------------------------------------------
extern "C" void kernel_launch(void* const* d_in, const int* in_sizes, int n_in,
                              void* d_out, int out_size, void* d_ws, size_t ws_size,
                              hipStream_t stream)
{
    const float* x  = (const float*)d_in[0];   // [32, 2048, 256] f32
    const float* W0 = (const float*)d_in[1];   // [512, 1024] f32
    const float* b0 = (const float*)d_in[2];   // [1024] f32
    const float* W1 = (const float*)d_in[3];   // [512, 1024] f32
    const float* b1 = (const float*)d_in[4];   // [1024] f32

    char* ws = (char*)d_ws;
    __bf16*   Wt0   = (__bf16*)ws;                                   // 1 MiB
    __bf16*   Wt1   = (__bf16*)(ws + (1u << 20));                    // 1 MiB
    __bf16*   h0seq = (__bf16*)(ws + (2u << 20));                    // 32 MiB [T][B][H]
    __bf16*   hbufs = (__bf16*)(ws + (2u << 20) + (32u << 20));      // 4 x 16 KiB
    unsigned* cnts  = (unsigned*)(ws + (2u << 20) + (32u << 20) + (64u << 10));

    lstm_prep_kernel<<<2048, 256, 0, stream>>>(W0, W1, Wt0, Wt1, hbufs, cnts);

    const int T = 2048;
    // layer 0: x f32 [B,T,256] -> h0seq bf16 [T,B,256]
    lstm_layer_kernel<<<8, 256, 0, stream>>>(
        x, /*in_is_f32=*/1, /*in_bs=*/(long)T * 256, /*in_ts=*/256L,
        Wt0, b0,
        h0seq, /*out_is_f32=*/0, /*out_bs=*/256L, /*out_ts=*/(long)32 * 256,
        hbufs, cnts + 0, T);

    // layer 1: h0seq bf16 [T,B,256] -> d_out f32 [B*T, 256]
    lstm_layer_kernel<<<8, 256, 0, stream>>>(
        h0seq, /*in_is_f32=*/0, /*in_bs=*/256L, /*in_ts=*/(long)32 * 256,
        Wt1, b1,
        d_out, /*out_is_f32=*/1, /*out_bs=*/(long)T * 256, /*out_ts=*/256L,
        hbufs + 2 * 32 * 256, cnts + 1, T);
}
